// LSTMModel_54683523612890
// MI455X (gfx1250) — matrix operand order repaired and checked by reading, not compiled, not run on hardware
//
#include <hip/hip_runtime.h>

typedef __attribute__((ext_vector_type(16))) __bf16        v16bf;
typedef __attribute__((ext_vector_type(8)))  float         v8f;
typedef __attribute__((ext_vector_type(4)))  unsigned int  u32x4;

#define EDIM 1024
#define HID  1024
#define VOC  32000
#define BSZ  32
#define SEQ  128

// ---------- helpers ----------

__device__ __forceinline__ unsigned short f2bf(float f) {
  unsigned int u = __float_as_uint(f);
  u += 0x7FFFu + ((u >> 16) & 1u);           // round-to-nearest-even
  return (unsigned short)(u >> 16);
}

__device__ __forceinline__ v8f zero8() {
  v8f z = {0.f, 0.f, 0.f, 0.f, 0.f, 0.f, 0.f, 0.f};
  return z;
}

union FragBF { v16bf v; u32x4 q[2]; };

// A-fragment (16x32 bf16 tile): two b128 loads at +0 and +32 bytes (K and K+16)
#define LDA(f, p) do { \
    (f).q[0] = *(const u32x4*)(p); \
    (f).q[1] = *(const u32x4*)((p) + 16); \
  } while (0)
// B-fragment (fragment-major, 512 contiguous bf16): lane's 16 values at lane*16
#define LDB(f, p) do { \
    (f).q[0] = *(const u32x4*)(p); \
    (f).q[1] = *(const u32x4*)((p) + 8); \
  } while (0)

#define WMMA_BF16(A, Bm, C) \
  __builtin_amdgcn_wmma_f32_16x16x32_bf16(false, (A), false, (Bm), (short)0, (C), false, false)

// ---------- weight / input preparation ----------

// Pack 4 gate weight matrices [1024x1024] f32 -> one fragment-major bf16
// matrix K=1024, N=4096 (gate-major columns).
__global__ __launch_bounds__(256) void pack_gate_w(
    const float* __restrict__ W0, const float* __restrict__ W1,
    const float* __restrict__ W2, const float* __restrict__ W3,
    unsigned short* __restrict__ dst) {
  int idx = blockIdx.x * 256 + threadIdx.x;          // < 1024*4096
  int within  = idx & 511;
  int fragIdx = idx >> 9;
  int l = within >> 4, s = within & 15;
  int kt = fragIdx >> 8;                              // 256 n-tiles
  int nt = fragIdx & 255;
  int k = kt * 32 + ((s >> 3) << 4) + ((l >> 4) << 3) + (s & 7);
  int n = nt * 16 + (l & 15);
  int g = n >> 10, col = n & 1023;
  const float* W = (g == 0) ? W0 : (g == 1) ? W1 : (g == 2) ? W2 : W3;
  dst[idx] = f2bf(W[(size_t)k * HID + col]);
}

// Pack W_hq [1024 x 32000] f32 -> fragment-major bf16 (2000 n-tiles).
__global__ __launch_bounds__(256) void pack_whq(
    const float* __restrict__ src, unsigned short* __restrict__ dst) {
  int idx = blockIdx.x * 256 + threadIdx.x;          // < 1024*32000
  int within  = idx & 511;
  int fragIdx = idx >> 9;
  int l = within >> 4, s = within & 15;
  int kt = fragIdx / 2000;
  int nt = fragIdx % 2000;
  int k = kt * 32 + ((s >> 3) << 4) + ((l >> 4) << 3) + (s & 7);
  int n = nt * 16 + (l & 15);
  dst[idx] = f2bf(src[(size_t)k * VOC + n]);
}

// Gather embeddings: Xe[t][b][e] = bf16(emb[X[b][t]][e])
__global__ __launch_bounds__(256) void embed_gather(
    const int* __restrict__ X, const float* __restrict__ emb,
    unsigned short* __restrict__ Xe) {
  int idx = blockIdx.x * 256 + threadIdx.x;          // < 128*32*1024
  int t   = idx >> 15;
  int rem = idx & 32767;
  int b   = rem >> 10;
  int e   = rem & 1023;
  int tok = X[b * SEQ + t];
  Xe[idx] = f2bf(emb[(size_t)tok * EDIM + e]);
}

__global__ __launch_bounds__(256) void init_state(
    const float* __restrict__ H, const float* __restrict__ C,
    float* __restrict__ hf, float* __restrict__ cf,
    unsigned short* __restrict__ hb) {
  int idx = blockIdx.x * 256 + threadIdx.x;          // < 32*1024
  float h = H[idx];
  hf[idx] = h;
  cf[idx] = C[idx];
  hb[idx] = f2bf(h);
}

__global__ __launch_bounds__(256) void finalize_state(
    const float* __restrict__ hf, const float* __restrict__ cf,
    float* __restrict__ out) {
  int idx = blockIdx.x * 256 + threadIdx.x;          // < 32*1024
  size_t off = (size_t)SEQ * BSZ * VOC;
  out[off + idx] = hf[idx];
  out[off + (size_t)BSZ * HID + idx] = cf[idx];
}

// ---------- recurrent gate kernel ----------
// Grid: 64 blocks x 256 thr (8 waves). Block owns ONE 16-column n-tile (per
// gate). Wave w -> (mTile = w&1, K-slice kq = w>>1 in [0,4)). The combined
// K=2048 reduction (x@Wx for kq<2, h@Wh for kq>=2) is split 4 ways; partial
// 16x16 tiles are summed through LDS, then kq==0 waves do the wave-local
// sigmoid/tanh cell update. Fragment streams are double-buffered with
// explicit pointer stepping (constant strides + immediate offsets) and a
// peeled epilogue so loads for kt+1 overlap the 4 WMMAs of kt.
__global__ __launch_bounds__(256) void lstm_gates(
    const unsigned short* __restrict__ Xe,
    const unsigned short* __restrict__ Wx,     // frag-major K=1024,N=4096
    const unsigned short* __restrict__ Wh,     // frag-major K=1024,N=4096
    const float* __restrict__ bi, const float* __restrict__ bff,
    const float* __restrict__ bo, const float* __restrict__ bc,
    const unsigned short* __restrict__ h_in,
    unsigned short* __restrict__ h_out,
    float* __restrict__ h_f32, float* __restrict__ c_f32, int t) {
  __shared__ float red[8][4][256];             // [kq*2+mTile][gate][r*16+c] 32 KB

  const int lane  = threadIdx.x & 31;
  const int w     = threadIdx.x >> 5;
  const int mTile = w & 1;
  const int kq    = w >> 1;                    // 0..3 K-slices
  const int m0    = mTile * 16;
  const int nt    = blockIdx.x;                // 0..63

  const unsigned short* A  = (kq < 2) ? (Xe + (size_t)t * (BSZ * EDIM)) : h_in;
  const unsigned short* Wf = (kq < 2) ? Wx : Wh;
  const int ktBase = (kq & 1) * 16;

  const int m  = lane & 15;
  const int kb = (lane >> 4) ? 8 : 0;

  // running pointers: A steps 32 elems/kt, W steps one full kt-row (256 frags)
  const unsigned short* pa = A + (size_t)(m0 + m) * HID + ktBase * 32 + kb;
  const unsigned short* pw = Wf + (size_t)ktBase * (256 * 512) +
                             (size_t)nt * 512 + lane * 16;
  // gate fragments live at immediate offsets from pw (64/128/192 n-tiles)
  const int GOF = 64 * 512;                    // elements between gate blocks
  const int WSTEP = 256 * 512;                 // elements per kt step
  const int ASTEP = 32;

  v8f aI = zero8(), aF = zero8(), aO = zero8(), aC = zero8();

  FragBF fa, b0, b1, b2, b3;
  LDA(fa, pa);
  LDB(b0, pw);
  LDB(b1, pw + GOF);
  LDB(b2, pw + 2 * GOF);
  LDB(b3, pw + 3 * GOF);

  for (int i = 0; i < 15; ++i) {
    pa += ASTEP;
    pw += WSTEP;
    FragBF nfa, nb0, nb1, nb2, nb3;
    LDA(nfa, pa);
    LDB(nb0, pw);
    LDB(nb1, pw + GOF);
    LDB(nb2, pw + 2 * GOF);
    LDB(nb3, pw + 3 * GOF);
    aI = WMMA_BF16(fa.v, b0.v, aI);
    aF = WMMA_BF16(fa.v, b1.v, aF);
    aO = WMMA_BF16(fa.v, b2.v, aO);
    aC = WMMA_BF16(fa.v, b3.v, aC);
    fa = nfa; b0 = nb0; b1 = nb1; b2 = nb2; b3 = nb3;
  }
  aI = WMMA_BF16(fa.v, b0.v, aI);
  aF = WMMA_BF16(fa.v, b1.v, aF);
  aO = WMMA_BF16(fa.v, b2.v, aO);
  aC = WMMA_BF16(fa.v, b3.v, aC);

  // dump partial tiles (C/D layout: VGPR r -> row rb+r, col lane&15)
  const int col = lane & 15;
  const int rb  = (lane >> 4) << 3;
  const int part = kq * 2 + mTile;
#pragma unroll
  for (int r = 0; r < 8; ++r) {
    red[part][0][(rb + r) * 16 + col] = aI[r];
    red[part][1][(rb + r) * 16 + col] = aF[r];
    red[part][2][(rb + r) * 16 + col] = aO[r];
    red[part][3][(rb + r) * 16 + col] = aC[r];
  }
  __syncthreads();

  if (kq == 0) {
    const int j = nt * 16 + col;
    const float biv = bi[j], bfv = bff[j], bov = bo[j], bcv = bc[j];
#pragma unroll
    for (int r = 0; r < 8; ++r) {
      const int e = (rb + r) * 16 + col;
      float sI = red[mTile][0][e] + red[2 + mTile][0][e] +
                 red[4 + mTile][0][e] + red[6 + mTile][0][e];
      float sF = red[mTile][1][e] + red[2 + mTile][1][e] +
                 red[4 + mTile][1][e] + red[6 + mTile][1][e];
      float sO = red[mTile][2][e] + red[2 + mTile][2][e] +
                 red[4 + mTile][2][e] + red[6 + mTile][2][e];
      float sC = red[mTile][3][e] + red[2 + mTile][3][e] +
                 red[4 + mTile][3][e] + red[6 + mTile][3][e];
      float iv = 1.f / (1.f + __expf(-(sI + biv)));
      float fv = 1.f / (1.f + __expf(-(sF + bfv)));
      float ov = 1.f / (1.f + __expf(-(sO + bov)));
      float cv = tanhf(sC + bcv);
      const int M = m0 + rb + r;
      size_t idx = (size_t)M * HID + j;
      float cn = fv * c_f32[idx] + iv * cv;
      float hn = ov * tanhf(cn);
      c_f32[idx] = cn;
      h_f32[idx] = hn;
      h_out[idx] = f2bf(hn);
    }
  }
}

// ---------- vocab projection kernel ----------
// Grid: 250 blocks x 256 thr. Block owns 128 vocab columns; wave w owns one
// 16-col n-tile for BOTH m-tiles (one B fragment feeds two WMMAs -> halves
// L2 weight traffic). h_t (64 KB bf16) staged in LDS; A fragments via
// ds_load_b128. K-loop double-buffered with pointer stepping + peeled tail.
__global__ __launch_bounds__(256) void lstm_proj(
    const unsigned short* __restrict__ h_in,   // [32][1024] bf16
    const unsigned short* __restrict__ Wq,     // frag-major K=1024,N=32000
    const float* __restrict__ bq,
    float* __restrict__ Y, int t) {
  __shared__ unsigned short sh[BSZ * HID];     // 64 KB
  {
    u32x4* d = (u32x4*)sh;
    const u32x4* s = (const u32x4*)h_in;
    for (int i = threadIdx.x; i < (BSZ * HID) / 8; i += 256) d[i] = s[i];
  }
  __syncthreads();

  const int lane = threadIdx.x & 31;
  const int w    = threadIdx.x >> 5;
  const int nt   = blockIdx.x * 8 + w;         // 0..1999
  const int m    = lane & 15;
  const int kb   = (lane >> 4) ? 8 : 0;

  const unsigned short* ps0 = sh + m * HID + kb;          // m-tile 0 row
  const unsigned short* ps1 = sh + (16 + m) * HID + kb;   // m-tile 1 row
  const unsigned short* pw  = Wq + (size_t)nt * 512 + lane * 16;
  const int ASTEP = 32;
  const int WSTEP = 2000 * 512;                // elements per kt step

  v8f acc0 = zero8(), acc1 = zero8();

  FragBF fa0, fa1, fb;
  LDA(fa0, ps0);
  LDA(fa1, ps1);
  LDB(fb, pw);

  for (int kt = 0; kt < 31; ++kt) {
    ps0 += ASTEP;
    ps1 += ASTEP;
    pw  += WSTEP;
    FragBF na0, na1, nb;
    LDA(na0, ps0);
    LDA(na1, ps1);
    LDB(nb, pw);
    acc0 = WMMA_BF16(fa0.v, fb.v, acc0);
    acc1 = WMMA_BF16(fa1.v, fb.v, acc1);
    fa0 = na0; fa1 = na1; fb = nb;
  }
  acc0 = WMMA_BF16(fa0.v, fb.v, acc0);
  acc1 = WMMA_BF16(fa1.v, fb.v, acc1);

  const int n  = nt * 16 + (lane & 15);
  const int rb = (lane >> 4) << 3;
  const float bqv = bq[n];
  size_t base = (size_t)t * BSZ * VOC;
#pragma unroll
  for (int r = 0; r < 8; ++r) {
    Y[base + (size_t)(rb + r) * VOC + n]      = acc0[r] + bqv;
    Y[base + (size_t)(16 + rb + r) * VOC + n] = acc1[r] + bqv;
  }
}

// ---------- launch ----------

extern "C" void kernel_launch(void* const* d_in, const int* in_sizes, int n_in,
                              void* d_out, int out_size, void* d_ws, size_t ws_size,
                              hipStream_t stream) {
  (void)in_sizes; (void)n_in; (void)out_size; (void)ws_size;
  const int*   X    = (const int*)d_in[0];
  const float* H    = (const float*)d_in[1];
  const float* C    = (const float*)d_in[2];
  const float* emb  = (const float*)d_in[3];
  const float* Wxi  = (const float*)d_in[4];
  const float* Whi  = (const float*)d_in[5];
  const float* bi   = (const float*)d_in[6];
  const float* Wxf  = (const float*)d_in[7];
  const float* Whf  = (const float*)d_in[8];
  const float* bff  = (const float*)d_in[9];
  const float* Wxo  = (const float*)d_in[10];
  const float* Who  = (const float*)d_in[11];
  const float* bo   = (const float*)d_in[12];
  const float* Wxc  = (const float*)d_in[13];
  const float* Whc  = (const float*)d_in[14];
  const float* bc   = (const float*)d_in[15];
  const float* Whq  = (const float*)d_in[16];
  const float* bq   = (const float*)d_in[17];
  float* out = (float*)d_out;

  char* ws = (char*)d_ws;
  unsigned short* Xe  = (unsigned short*)ws; ws += (size_t)SEQ * BSZ * EDIM * 2;  // 8.4 MB
  unsigned short* WxF = (unsigned short*)ws; ws += (size_t)EDIM * 4 * HID * 2;    // 8.4 MB
  unsigned short* WhF = (unsigned short*)ws; ws += (size_t)HID  * 4 * HID * 2;    // 8.4 MB
  unsigned short* WqF = (unsigned short*)ws; ws += (size_t)HID  * VOC * 2;        // 65.5 MB
  unsigned short* hb0 = (unsigned short*)ws; ws += (size_t)BSZ * HID * 2;
  unsigned short* hb1 = (unsigned short*)ws; ws += (size_t)BSZ * HID * 2;
  float* hf = (float*)ws; ws += (size_t)BSZ * HID * 4;
  float* cf = (float*)ws; ws += (size_t)BSZ * HID * 4;

  pack_gate_w<<<16384, 256, 0, stream>>>(Wxi, Wxf, Wxo, Wxc, WxF);
  pack_gate_w<<<16384, 256, 0, stream>>>(Whi, Whf, Who, Whc, WhF);
  pack_whq<<<128000, 256, 0, stream>>>(Whq, WqF);
  embed_gather<<<16384, 256, 0, stream>>>(X, emb, Xe);
  init_state<<<128, 256, 0, stream>>>(H, C, hf, cf, hb0);

  for (int t = 0; t < SEQ; ++t) {
    unsigned short* hin  = (t & 1) ? hb1 : hb0;
    unsigned short* hout = (t & 1) ? hb0 : hb1;
    lstm_gates<<<64, 256, 0, stream>>>(Xe, WxF, WhF, bi, bff, bo, bc,
                                       hin, hout, hf, cf, t);
    lstm_proj<<<250, 256, 0, stream>>>(hout, WqF, bq, out, t);
  }
  finalize_state<<<128, 256, 0, stream>>>(hf, cf, out);
}
